// DecoderLSTM_42898133352867
// MI455X (gfx1250) — matrix operand order repaired and checked by reading, not compiled, not run on hardware
//
#include <hip/hip_runtime.h>

#define Hs 512
#define Es 512
#define BATCH 32
#define SEQ 128
#define VOCAB 50257
#define G4H 2048      // 4*H
#define KC0 1536      // E + H(feed) + H(h0)  : layer0 concat K
#define KC1 1024      // H(h0) + H(h1)        : layer1 concat K

#define PBLK 128      // projection output block: 128x128 per workgroup
#define LDSTR 40      // LDS row stride in elements (32 + 8 pad, 80B, 16B-aligned)

typedef __attribute__((ext_vector_type(16))) __bf16 bf16x16;
typedef __attribute__((ext_vector_type(8)))  __bf16 bf16x8;
typedef __attribute__((ext_vector_type(8)))  float  f32x8;

__device__ __forceinline__ unsigned short f32_to_bf16_bits(float f) {
  unsigned int u = __float_as_uint(f);
  u += 0x7FFFu + ((u >> 16) & 1u);   // round-to-nearest-even
  return (unsigned short)(u >> 16);
}

__device__ __forceinline__ float sigmoidf_(float x) {
  return 1.0f / (1.0f + __expf(-x));
}

// ---------------- WMMA tile loaders (contiguous k order, wave32) --------------
// A tile: 16x32 bf16, row-major source with leading dim lda (elements).
// lane<16: row=lane,     K = {k0+0..15}
// lane>=16: row=lane-16, K = {k0+16..31}
// Only 16B alignment assumed (two bf16x8 chunks). Works on global or LDS ptrs.
__device__ __forceinline__ bf16x16 load_a_tile(const unsigned short* A, int lda,
                                               int m0, int k0) {
  int lane = threadIdx.x & 31;
  const unsigned short* p =
      A + (size_t)(m0 + (lane & 15)) * lda + k0 + ((lane >> 4) << 4);
  bf16x8 lo = *(const bf16x8*)p;
  bf16x8 hi = *(const bf16x8*)(p + 8);
  bf16x16 a;
#pragma unroll
  for (int i = 0; i < 8; ++i) { a[i] = lo[i]; a[i + 8] = hi[i]; }
  return a;
}

// B tile: 32x16 bf16 where B = W^T and W is row-major [N x K] (ldw = K).
// lane holds column n0+(lane&15); K = k0 + (lane>=16 ? 16 : 0) + e.
// Global version: 32B-aligned single load (rows are 1KB+ aligned).
__device__ __forceinline__ bf16x16 load_bt_tile(const unsigned short* W, int ldw,
                                                int n0, int k0, int nmax) {
  int lane = threadIdx.x & 31;
  int col = n0 + (lane & 15);
  if (col >= nmax) col = nmax - 1;   // clamp; stores are guarded separately
  const unsigned short* p = W + (size_t)col * ldw + k0 + ((lane >> 4) << 4);
  return *(const bf16x16*)p;
}

// LDS version of B-frag loader: 16B-aligned chunks only (stride 80B).
__device__ __forceinline__ bf16x16 lds_bt_frag(const unsigned short* B, int n0) {
  int lane = threadIdx.x & 31;
  const unsigned short* p =
      B + (size_t)(n0 + (lane & 15)) * LDSTR + ((lane >> 4) << 4);
  bf16x8 lo = *(const bf16x8*)p;
  bf16x8 hi = *(const bf16x8*)(p + 8);
  bf16x16 b;
#pragma unroll
  for (int i = 0; i < 8; ++i) { b[i] = lo[i]; b[i + 8] = hi[i]; }
  return b;
}

// ---------------- LSTM gate GEMM: C[32 x 2048] = A[32 x K] * W[2048 x K]^T + bias
__global__ void lstm_gate_gemm(const unsigned short* __restrict__ A,
                               const unsigned short* __restrict__ W,
                               const float* __restrict__ bias,
                               float* __restrict__ C, int K) {
  int n0 = blockIdx.x << 4;
  int m0 = blockIdx.y << 4;
  f32x8 acc = {};
  for (int k = 0; k < K; k += 32) {
    bf16x16 a = load_a_tile(A, K, m0, k);
    bf16x16 b = load_bt_tile(W, K, n0, k, G4H);
    acc = __builtin_amdgcn_wmma_f32_16x16x32_bf16(false, a, false, b,
                                                  (short)0, acc, false, false);
  }
  int lane = threadIdx.x & 31;
  int col = n0 + (lane & 15);
  int rbase = m0 + ((lane >> 4) << 3);
  float bv = bias[col];
#pragma unroll
  for (int r = 0; r < 8; ++r)
    C[(size_t)(rbase + r) * G4H + col] = acc[r] + bv;
}

// ---------------- Output projection, LDS-tiled ----------------
// out[4096 x V] = A[4096 x 512] * Wf[V x 512]^T + bf
// 128x128 output block / workgroup (256 thr = 8 waves).
// Per k-step: stage A[128x32] + Wf-rows[128x32] in LDS; each wave computes
// 2 M-tiles x 4 N-tiles (B frags reused 2x, A frags 4x, LDS shared 8 ways).
// L2 traffic: (50272/128)*4MB + (4096/128)*51MB ~= 3.2GB (vs 16GB unblocked).
__global__ __launch_bounds__(256) void proj_gemm_lds(
    const unsigned short* __restrict__ A,
    const unsigned short* __restrict__ Wf,
    const float* __restrict__ bfv,
    float* __restrict__ out) {
  __shared__ unsigned short ldsA[PBLK * LDSTR];
  __shared__ unsigned short ldsB[PBLK * LDSTR];

  int nblock = blockIdx.x * PBLK;
  int mblock = blockIdx.y * PBLK;
  int t = threadIdx.x;
  int lane = t & 31;
  int w = t >> 5;           // wave 0..7
  int wrow = w >> 1;        // 0..3 -> M-tiles {wrow*2, wrow*2+1}
  int wcol = w & 1;         // 0..1 -> N-tiles {wcol*4 .. wcol*4+3}

  // staging assignment: thread t copies 32B of one row
  int crow = t >> 1;              // 0..127
  int chalf = (t & 1) << 4;       // 0 or 16 elements
  int brow = nblock + crow;
  if (brow >= VOCAB) brow = VOCAB - 1;          // clamped; stores guarded
  const uint4* gA = (const uint4*)(A + (size_t)(mblock + crow) * Hs);
  const uint4* gB = (const uint4*)(Wf + (size_t)brow * Hs);
  uint4* lA = (uint4*)(ldsA + crow * LDSTR + chalf);
  uint4* lB = (uint4*)(ldsB + crow * LDSTR + chalf);

  f32x8 acc[2][4] = {};

  for (int k = 0; k < Hs; k += 32) {
    int q = (k + chalf) >> 3;     // uint4 index (8 bf16 per uint4)
    lA[0] = gA[q];
    lA[1] = gA[q + 1];
    lB[0] = gB[q];
    lB[1] = gB[q + 1];
    __syncthreads();

    bf16x16 a0 = load_a_tile(ldsA, LDSTR, wrow * 32,      0);
    bf16x16 a1 = load_a_tile(ldsA, LDSTR, wrow * 32 + 16, 0);
#pragma unroll
    for (int j = 0; j < 4; ++j) {
      bf16x16 b = lds_bt_frag(ldsB, (wcol * 4 + j) * 16);
      acc[0][j] = __builtin_amdgcn_wmma_f32_16x16x32_bf16(false, a0, false, b,
                                                          (short)0, acc[0][j], false, false);
      acc[1][j] = __builtin_amdgcn_wmma_f32_16x16x32_bf16(false, a1, false, b,
                                                          (short)0, acc[1][j], false, false);
    }
    __syncthreads();
  }

  int r0 = (lane >> 4) << 3;
#pragma unroll
  for (int j = 0; j < 4; ++j) {
    int col = nblock + (wcol * 4 + j) * 16 + (lane & 15);
    if (col < VOCAB) {
      float bv = bfv[col];
#pragma unroll
      for (int i = 0; i < 2; ++i) {
        int rbase = mblock + (wrow * 2 + i) * 16 + r0;
#pragma unroll
        for (int r = 0; r < 8; ++r)
          out[(size_t)(rbase + r) * VOCAB + col] = acc[i][j][r] + bv;
      }
    }
  }
}

// ---------------- Prep kernels ----------------
__global__ void conv_wcat0(const float* __restrict__ Wih, const float* __restrict__ Whh,
                           unsigned short* __restrict__ Wcat) {
  int idx = blockIdx.x * blockDim.x + threadIdx.x;
  if (idx >= G4H * KC0) return;
  int row = idx / KC0, col = idx % KC0;
  float v = (col < 1024) ? Wih[(size_t)row * 1024 + col]
                         : Whh[(size_t)row * 512 + (col - 1024)];
  Wcat[idx] = f32_to_bf16_bits(v);
}

__global__ void conv_wcat1(const float* __restrict__ Wih, const float* __restrict__ Whh,
                           unsigned short* __restrict__ Wcat) {
  int idx = blockIdx.x * blockDim.x + threadIdx.x;
  if (idx >= G4H * KC1) return;
  int row = idx / KC1, col = idx % KC1;
  float v = (col < 512) ? Wih[(size_t)row * 512 + col]
                        : Whh[(size_t)row * 512 + (col - 512)];
  Wcat[idx] = f32_to_bf16_bits(v);
}

__global__ void conv_bf16(const float* __restrict__ src, unsigned short* __restrict__ dst, int n) {
  int idx = blockIdx.x * blockDim.x + threadIdx.x;
  if (idx < n) dst[idx] = f32_to_bf16_bits(src[idx]);
}

__global__ void sum_bias(const float* __restrict__ a, const float* __restrict__ b,
                         float* __restrict__ o, int n) {
  int idx = blockIdx.x * blockDim.x + threadIdx.x;
  if (idx < n) o[idx] = a[idx] + b[idx];
}

// x part of A0_all[s][b][0:512] = bf16(emb[seq[b][s]])
__global__ void build_x(const int* __restrict__ seq, const float* __restrict__ emb,
                        unsigned short* __restrict__ A0all) {
  int idx = blockIdx.x * blockDim.x + threadIdx.x;
  if (idx >= SEQ * BATCH * Es) return;
  int e = idx & 511;
  int b = (idx >> 9) & 31;
  int s = idx >> 14;
  int tok = seq[b * SEQ + s];
  A0all[((size_t)s * BATCH + b) * KC0 + e] = f32_to_bf16_bits(emb[(size_t)tok * Es + e]);
}

__global__ void init_state(const float* __restrict__ enc_h, const float* __restrict__ enc_c,
                           unsigned short* __restrict__ A0all, unsigned short* __restrict__ A1,
                           float* __restrict__ c0, float* __restrict__ c1) {
  int idx = blockIdx.x * blockDim.x + threadIdx.x;
  if (idx >= BATCH * Hs) return;
  int j = idx & 511, b = idx >> 9;
  A0all[(size_t)b * KC0 + 512 + j]  = 0;                                   // feed0 = 0
  A0all[(size_t)b * KC0 + 1024 + j] = f32_to_bf16_bits(enc_h[idx]);        // h0 init
  A1[(size_t)b * KC1 + 512 + j]     = f32_to_bf16_bits(enc_h[BATCH * Hs + idx]); // h1 init
  c0[idx] = enc_c[idx];
  c1[idx] = enc_c[BATCH * Hs + idx];
}

// ---------------- Cell (elementwise) kernels ----------------
__global__ void cell0(const float* __restrict__ g, float* __restrict__ c0,
                      unsigned short* __restrict__ A1, unsigned short* __restrict__ A0next) {
  int idx = blockIdx.x * blockDim.x + threadIdx.x;
  if (idx >= BATCH * Hs) return;
  int j = idx & 511, b = idx >> 9;
  const float* gb = g + (size_t)b * G4H;
  float i  = sigmoidf_(gb[j]);
  float f  = sigmoidf_(gb[512 + j]);
  float gg = tanhf(gb[1024 + j]);
  float o  = sigmoidf_(gb[1536 + j]);
  float c2 = f * c0[idx] + i * gg;
  c0[idx] = c2;
  float h = o * tanhf(c2);
  unsigned short hb = f32_to_bf16_bits(h);
  A1[(size_t)b * KC1 + j] = hb;                              // layer1 input slot
  if (A0next) A0next[(size_t)b * KC0 + 1024 + j] = hb;       // next step h0 slot
}

__global__ void cell1(const float* __restrict__ g, float* __restrict__ c1,
                      unsigned short* __restrict__ outsA, int s,
                      unsigned short* __restrict__ A1, unsigned short* __restrict__ A0next) {
  int idx = blockIdx.x * blockDim.x + threadIdx.x;
  if (idx >= BATCH * Hs) return;
  int j = idx & 511, b = idx >> 9;
  const float* gb = g + (size_t)b * G4H;
  float i  = sigmoidf_(gb[j]);
  float f  = sigmoidf_(gb[512 + j]);
  float gg = tanhf(gb[1024 + j]);
  float o  = sigmoidf_(gb[1536 + j]);
  float c2 = f * c1[idx] + i * gg;
  c1[idx] = c2;
  float h = o * tanhf(c2);
  unsigned short hb = f32_to_bf16_bits(h);
  outsA[((size_t)b * SEQ + s) * Hs + j] = hb;                // projection A, row b*S+s
  A1[(size_t)b * KC1 + 512 + j] = hb;                        // h1 slot for next gemm1
  if (A0next) A0next[(size_t)b * KC0 + 512 + j] = hb;        // next step feed slot
}

// ---------------- Host launcher ----------------
extern "C" void kernel_launch(void* const* d_in, const int* in_sizes, int n_in,
                              void* d_out, int out_size, void* d_ws, size_t ws_size,
                              hipStream_t stream) {
  (void)in_sizes; (void)n_in; (void)out_size; (void)ws_size;
  const int*   seq   = (const int*)d_in[0];
  const float* enc_h = (const float*)d_in[1];
  const float* enc_c = (const float*)d_in[2];
  const float* emb   = (const float*)d_in[3];
  const float* Wih0  = (const float*)d_in[4];
  const float* Whh0  = (const float*)d_in[5];
  const float* bih0  = (const float*)d_in[6];
  const float* bhh0  = (const float*)d_in[7];
  const float* Wih1  = (const float*)d_in[8];
  const float* Whh1  = (const float*)d_in[9];
  const float* bih1  = (const float*)d_in[10];
  const float* bhh1  = (const float*)d_in[11];
  const float* Wf    = (const float*)d_in[12];
  const float* bfv   = (const float*)d_in[13];
  float* out = (float*)d_out;

  char* p = (char*)d_ws;
  auto carve = [&](size_t bytes) -> void* {
    void* r = (void*)p;
    p += (bytes + 255) & ~(size_t)255;
    return r;
  };
  unsigned short* Wcat0 = (unsigned short*)carve((size_t)G4H * KC0 * 2);
  unsigned short* Wcat1 = (unsigned short*)carve((size_t)G4H * KC1 * 2);
  unsigned short* WfB   = (unsigned short*)carve((size_t)VOCAB * Hs * 2);
  unsigned short* A0all = (unsigned short*)carve((size_t)SEQ * BATCH * KC0 * 2);
  unsigned short* A1b   = (unsigned short*)carve((size_t)BATCH * KC1 * 2);
  unsigned short* outsA = (unsigned short*)carve((size_t)BATCH * SEQ * Hs * 2);
  float* gates0 = (float*)carve((size_t)BATCH * G4H * 4);
  float* gates1 = (float*)carve((size_t)BATCH * G4H * 4);
  float* c0     = (float*)carve((size_t)BATCH * Hs * 4);
  float* c1     = (float*)carve((size_t)BATCH * Hs * 4);
  float* bsum0  = (float*)carve((size_t)G4H * 4);
  float* bsum1  = (float*)carve((size_t)G4H * 4);

  // ---- prep: weight conversions, bias sums, embedding gather, state init
  { int n = G4H * KC0; conv_wcat0<<<(n + 255) / 256, 256, 0, stream>>>(Wih0, Whh0, Wcat0); }
  { int n = G4H * KC1; conv_wcat1<<<(n + 255) / 256, 256, 0, stream>>>(Wih1, Whh1, Wcat1); }
  { int n = VOCAB * Hs; conv_bf16<<<(n + 255) / 256, 256, 0, stream>>>(Wf, WfB, n); }
  sum_bias<<<(G4H + 255) / 256, 256, 0, stream>>>(bih0, bhh0, bsum0, G4H);
  sum_bias<<<(G4H + 255) / 256, 256, 0, stream>>>(bih1, bhh1, bsum1, G4H);
  { int n = SEQ * BATCH * Es; build_x<<<(n + 255) / 256, 256, 0, stream>>>(seq, emb, A0all); }
  init_state<<<(BATCH * Hs + 255) / 256, 256, 0, stream>>>(enc_h, enc_c, A0all, A1b, c0, c1);

  // ---- sequential recurrence: 128 steps, 4 launches each, ordered by stream
  dim3 ggrid(G4H / 16, BATCH / 16);   // 128 x 2 tiles, one wave per 16x16 tile
  int cellBlocks = (BATCH * Hs + 255) / 256;
  for (int s = 0; s < SEQ; ++s) {
    unsigned short* A0s = A0all + (size_t)s * BATCH * KC0;
    unsigned short* A0n = (s + 1 < SEQ) ? A0all + (size_t)(s + 1) * BATCH * KC0 : nullptr;
    lstm_gate_gemm<<<ggrid, 32, 0, stream>>>(A0s, Wcat0, bsum0, gates0, KC0);
    cell0<<<cellBlocks, 256, 0, stream>>>(gates0, c0, A1b, A0n);
    lstm_gate_gemm<<<ggrid, 32, 0, stream>>>(A1b, Wcat1, bsum1, gates1, KC1);
    cell1<<<cellBlocks, 256, 0, stream>>>(gates1, c1, outsA, s, A1b, A0n);
  }

  // ---- final projection: LDS-tiled 128x128 blocks (store-bound, ~824 MB out)
  dim3 pgrid((VOCAB + PBLK - 1) / PBLK, (BATCH * SEQ) / PBLK);  // 393 x 32
  proj_gemm_lds<<<pgrid, 256, 0, stream>>>(outsA, WfB, bfv, out);
}
